// LocalTransformer_54314156425832
// MI455X (gfx1250) — matrix operand order repaired and checked by reading, not compiled, not run on hardware
//
#include <hip/hip_runtime.h>
#include <math.h>

typedef __bf16 bf16_t;
typedef __bf16  v16bf __attribute__((ext_vector_type(16)));
typedef float   v8f   __attribute__((ext_vector_type(8)));

#define BATCH 4
#define SEQ   2048
#define DIMM  512
#define HEADS 8
#define DH    64
#define WWIN  256
#define NWIN  8
#define MTOK  (BATCH * SEQ)      // 8192
#define FFI      1365
#define FFI2     2730
#define FFI2PAD  2752            // 43 * 64
#define FFIPAD   1376            // 43 * 32
#define NEGINF  (-3.4028235e38f)

// ---------------------------------------------------------------------------
// WMMA fragment loaders (16x16x32 bf16; contiguous k order, same for A and B)
// A (16x32, row-major source): lane m=lane&15, khalf=lane>>4;
//   e[0..15]=A[m][kk+khalf*16 .. +15]
// B (32x16, from transposed source T[n][k]): lane col=n0+(lane&15),
//   16 consecutive K values starting at kk + (lane>>4)*16
// ---------------------------------------------------------------------------
__device__ inline v16bf load_a_frag(const bf16_t* A, int lda, int m, int kk, int lane) {
  int khalf = (lane >> 4) & 1;
  const bf16_t* p = A + (size_t)m * lda + kk + khalf * 16;
  union { uint4 u[2]; v16bf v; } r;
  r.u[0] = *(const uint4*)p;
  r.u[1] = *(const uint4*)(p + 8);
  return r.v;
}

__device__ inline v16bf load_b_frag(const bf16_t* WT, int ldwt, int n, int kk, int lane) {
  int col = n + (lane & 15);
  int kb  = kk + ((lane >> 4) & 1) * 16;
  const bf16_t* p = WT + (size_t)col * ldwt + kb;
  union { uint4 u[2]; v16bf v; } r;
  r.u[0] = *(const uint4*)p;
  r.u[1] = *(const uint4*)(p + 8);
  return r.v;
}

// ---------------------------------------------------------------------------
// Generic bf16 WMMA GEMM: C[M,N] = A[M,K] * W[K,N] (+bias) (+residual)
// A row-major bf16 (lda), W provided transposed: WT[N_pad, K_pad] bf16 (ldwt).
// 256 threads = 8 waves; each wave: 16 rows x 64 cols. Grid: (M/128, ceil(N/64))
// QKV_MODE: cols [0,1024) -> Cb (q,k row-major); cols [1024,1536) -> Vt
//           stored transposed as Vt[b][h][d][token] for fast P*V B-fragments.
// ---------------------------------------------------------------------------
template<bool HAS_BIAS, bool HAS_RES, bool OUT_F32, bool OUT_BF16, bool QKV_MODE>
__global__ void __launch_bounds__(256)
gemm_bf16_k(const bf16_t* __restrict__ A, int lda,
            const bf16_t* __restrict__ WT, int ldwt,
            const float* __restrict__ bias,
            const float* res, int ldres,
            float* Cf, int ldcf,
            bf16_t* __restrict__ Cb, int ldcb,
            bf16_t* __restrict__ Vt,
            int M, int K, int nlimit)
{
  const int lane = threadIdx.x & 31;
  const int wv   = threadIdx.x >> 5;
  const int m0   = blockIdx.x * 128 + wv * 16;
  const int n0   = blockIdx.y * 64;
  if (m0 >= M) return;

  v8f z = {0.f,0.f,0.f,0.f,0.f,0.f,0.f,0.f};
  v8f acc[4] = { z, z, z, z };

  for (int kk = 0; kk < K; kk += 32) {
    v16bf af = load_a_frag(A, lda, m0 + (lane & 15), kk, lane);
#pragma unroll
    for (int t = 0; t < 4; ++t) {
      v16bf bfr = load_b_frag(WT, ldwt, n0 + t * 16, kk, lane);
      acc[t] = __builtin_amdgcn_wmma_f32_16x16x32_bf16(
          false, af, false, bfr, (short)0, acc[t], false, false);
    }
  }

  const int mrow = (lane >> 4) * 8;
  const int ncol = lane & 15;
#pragma unroll
  for (int t = 0; t < 4; ++t) {
    int n = n0 + t * 16 + ncol;
    if (n >= nlimit) continue;
    float bv = HAS_BIAS ? bias[n] : 0.f;
#pragma unroll
    for (int e = 0; e < 8; ++e) {
      int m = m0 + mrow + e;
      float v = acc[t][e] + bv;
      if (HAS_RES)  v += res[(size_t)m * ldres + n];
      if (OUT_F32)  Cf[(size_t)m * ldcf + n] = v;
      if (OUT_BF16) {
        if (QKV_MODE && n >= 1024) {        // V section -> transposed store
          int d  = (n - 1024) & 63;
          int hh = (n - 1024) >> 6;
          int bb = m >> 11;
          int tok = m & 2047;
          Vt[(((size_t)bb * HEADS + hh) * DH + d) * SEQ + tok] = (bf16_t)v;
        } else {
          Cb[(size_t)m * ldcb + n] = (bf16_t)v;
        }
      }
    }
  }
}

// ---------------------------------------------------------------------------
// Fused local attention, one block per (b, head, window, quarter).
// 4 waves x 16 query rows; S strip (16x512 f32) per wave in LDS (128 KB total)
// Window 0: first 256 keys are fully masked -> skip their tiles entirely.
// ---------------------------------------------------------------------------
__global__ void __launch_bounds__(128)
attn_k(const bf16_t* __restrict__ qkv, const bf16_t* __restrict__ vt,
       const float* __restrict__ biasT, bf16_t* __restrict__ o)
{
  extern __shared__ float smem[];
  const int lane = threadIdx.x & 31;
  const int wv   = threadIdx.x >> 5;
  const int blk  = blockIdx.x;          // ((b*NWIN + w)*HEADS + h)*4 + q
  const int qq = blk & 3;
  const int h  = (blk >> 2) & 7;
  const int w  = (blk >> 5) & 7;
  const int b  = (blk >> 8);
  const int r0 = qq * 64 + wv * 16;     // query row within window
  const int LD = 3 * DIMM;              // 1536
  float* S = smem + (size_t)wv * 16 * 512;
  const float scale = 0.125f;           // DH^-0.5
  const int mrow = (lane >> 4) * 8;
  const int ncol = lane & 15;

  // preload Q A-fragments (K = DH = 64 -> two k-steps)
  const bf16_t* Aq = qkv + ((size_t)(b * SEQ + w * WWIN + r0)) * LD + h * DH;
  v16bf aq[2];
  aq[0] = load_a_frag(Aq, LD, lane & 15, 0,  lane);
  aq[1] = load_a_frag(Aq, LD, lane & 15, 32, lane);

  const bf16_t* Kbase = qkv + (size_t)b * SEQ * LD + DIMM + h * DH;

  // ---- S = Q K^T -------------------------------------------------------
  const int jt0 = (w == 0) ? 16 : 0;    // fully-masked leading key tiles
  for (int jt = 0; jt < jt0; ++jt) {
    int j = jt * 16 + ncol;
#pragma unroll
    for (int e = 0; e < 8; ++e) S[(mrow + e) * 512 + j] = NEGINF;
  }
  for (int jt = jt0; jt < 32; ++jt) {
    v8f acc = {0.f,0.f,0.f,0.f,0.f,0.f,0.f,0.f};
    int j  = jt * 16 + ncol;
    int kt = (w - 1) * WWIN + j;        // >= 0 for all processed tiles
    const bf16_t* kp = Kbase + (size_t)kt * LD;
#pragma unroll
    for (int s = 0; s < 2; ++s) {
      int kb = s * 32 + ((lane >> 4) & 1) * 16;
      union { uint4 u[2]; v16bf v; } r;
      r.u[0] = *(const uint4*)(kp + kb);
      r.u[1] = *(const uint4*)(kp + kb + 8);
      acc = __builtin_amdgcn_wmma_f32_16x16x32_bf16(
          false, aq[s], false, r.v, (short)0, acc, false, false);
    }
#pragma unroll
    for (int e = 0; e < 8; ++e) {
      int i = r0 + mrow + e;            // query position in window
      float v = acc[e] * scale + biasT[((size_t)h * WWIN + i) * 512 + j];
      bool msk = (j < i) || (j > WWIN + i);
      S[(mrow + e) * 512 + j] = msk ? NEGINF : v;
    }
  }
  __syncthreads();

  // ---- softmax over 512 keys: lane = (row, half) -------------------------
  {
    int row  = lane & 15;
    int half = lane >> 4;
    float* Sr = S + row * 512 + half * 256;
    float mx = NEGINF;
    for (int c = 0; c < 256; ++c) mx = fmaxf(mx, Sr[c]);
    mx = fmaxf(mx, __shfl_xor(mx, 16));
    float sum = 0.f;
    for (int c = 0; c < 256; ++c) { float p = __expf(Sr[c] - mx); Sr[c] = p; sum += p; }
    sum += __shfl_xor(sum, 16);
    float inv = 1.f / sum;
    for (int c = 0; c < 256; ++c) Sr[c] *= inv;
  }
  __syncthreads();

  // ---- O = P V : A-frag built once per k-step, reused for 4 col tiles ----
  const bf16_t* Vh = vt + ((size_t)(b * HEADS + h)) * DH * SEQ;
  const int kk0 = (w == 0) ? WWIN : 0;
  v8f z = {0.f,0.f,0.f,0.f,0.f,0.f,0.f,0.f};
  v8f acc[4] = { z, z, z, z };
  for (int kk = kk0; kk < 512; kk += 32) {
    int m = lane & 15;
    int khalf = (lane >> 4) & 1;
    const float* p0 = S + m * 512 + kk + khalf * 16;
    v16bf af;
#pragma unroll
    for (int e = 0; e < 8; ++e) {
      af[e]     = (bf16_t)p0[e];
      af[e + 8] = (bf16_t)p0[8 + e];
    }
    int kt0 = (w - 1) * WWIN + kk + khalf * 16;   // >= 0 in processed range
#pragma unroll
    for (int dt = 0; dt < 4; ++dt) {
      const bf16_t* vrow = Vh + (size_t)(dt * 16 + ncol) * SEQ + kt0;
      union { uint4 u[2]; v16bf v; } r;
      r.u[0] = *(const uint4*)vrow;
      r.u[1] = *(const uint4*)(vrow + 8);
      acc[dt] = __builtin_amdgcn_wmma_f32_16x16x32_bf16(
          false, af, false, r.v, (short)0, acc[dt], false, false);
    }
  }
#pragma unroll
  for (int dt = 0; dt < 4; ++dt) {
#pragma unroll
    for (int e = 0; e < 8; ++e) {
      int i = r0 + mrow + e;
      size_t tok = (size_t)b * SEQ + w * WWIN + i;
      o[tok * DIMM + h * DH + dt * 16 + ncol] = (bf16_t)acc[dt][e];
    }
  }
}

// ---------------------------------------------------------------------------
// LayerNorm: one wave per token (512 elems, 16 per lane), output bf16
// ---------------------------------------------------------------------------
__global__ void __launch_bounds__(128)
ln_k(const float* __restrict__ x, const float* __restrict__ g,
     const float* __restrict__ bt, bf16_t* __restrict__ y, int Mtok)
{
  int lane = threadIdx.x & 31;
  int wv   = threadIdx.x >> 5;
  int m = blockIdx.x * 4 + wv;
  if (m >= Mtok) return;
  const float* xr = x + (size_t)m * DIMM;
  float vals[16];
  float s = 0.f;
#pragma unroll
  for (int k = 0; k < 16; ++k) { vals[k] = xr[lane + 32 * k]; s += vals[k]; }
  for (int off = 16; off; off >>= 1) s += __shfl_xor(s, off);
  float mu = s * (1.f / DIMM);
  float vs = 0.f;
#pragma unroll
  for (int k = 0; k < 16; ++k) { float d = vals[k] - mu; vs += d * d; }
  for (int off = 16; off; off >>= 1) vs += __shfl_xor(vs, off);
  float rs = rsqrtf(vs * (1.f / DIMM) + 1e-5f);
#pragma unroll
  for (int k = 0; k < 16; ++k) {
    int c = lane + 32 * k;
    y[(size_t)m * DIMM + c] = (bf16_t)((vals[k] - mu) * rs * g[c] + bt[c]);
  }
}

// ---------------------------------------------------------------------------
// time embedding MLP (B=4), plain f32 (negligible cost)
// ---------------------------------------------------------------------------
__global__ void __launch_bounds__(256)
time_k(const float* __restrict__ tim, const float* __restrict__ w1,
       const float* __restrict__ b1, const float* __restrict__ w2,
       const float* __restrict__ b2, float* __restrict__ temb)
{
  __shared__ float e[DIMM];
  __shared__ float y1[DIMM];
  int b = blockIdx.x, t = threadIdx.x;
  float tv = tim[b];
  for (int c = t; c < DIMM; c += 256) {
    int k = c & 255;
    float fr = __expf(-9.210340371976184f * (float)k / 256.f);
    float arg = tv * fr;
    e[c] = (c < 256) ? __cosf(arg) : __sinf(arg);
  }
  __syncthreads();
  for (int c = t; c < DIMM; c += 256) {
    float s = b1[c];
    for (int k = 0; k < DIMM; ++k) s += e[k] * w1[k * DIMM + c];
    y1[c] = s / (1.f + __expf(-s));         // silu
  }
  __syncthreads();
  for (int c = t; c < DIMM; c += 256) {
    float s = b2[c];
    for (int k = 0; k < DIMM; ++k) s += y1[k] * w2[k * DIMM + c];
    temb[b * DIMM + c] = s;
  }
}

// ---------------------------------------------------------------------------
// DPB MLP: one block per relative distance r (512 rows) -> bias_all[512][8]
// ---------------------------------------------------------------------------
__global__ void __launch_bounds__(256)
dpb_k(const float* __restrict__ w1, const float* __restrict__ b1,
      const float* __restrict__ w2, const float* __restrict__ b2,
      const float* __restrict__ w3, const float* __restrict__ b3,
      float* __restrict__ ball)
{
  __shared__ float d1[256];
  __shared__ float d2[256];
  int r = blockIdx.x, t = threadIdx.x;
  float s = (float)r * w1[t] + b1[t];
  d1[t] = s / (1.f + __expf(-s));
  __syncthreads();
  float s2 = b2[t];
  for (int k = 0; k < 256; ++k) s2 += d1[k] * w2[k * 256 + t];
  d2[t] = s2 / (1.f + __expf(-s2));
  __syncthreads();
  if (t < HEADS) {
    float s3 = b3[t];
    for (int k = 0; k < 256; ++k) s3 += d2[k] * w3[k * HEADS + t];
    ball[r * HEADS + t] = s3;
  }
}

__global__ void __launch_bounds__(256)
dpbbias_k(const float* __restrict__ ball, float* __restrict__ biasT)
{
  int idx = blockIdx.x * 256 + threadIdx.x;   // HEADS*256*512
  int hh = idx >> 17;
  int rem = idx & 131071;
  int i = rem >> 9, j = rem & 511;
  int r = 256 + i - j; if (r < 0) r = -r;
  biasT[idx] = ball[r * HEADS + hh];
}

// ---------------------------------------------------------------------------
// elementwise helpers
// ---------------------------------------------------------------------------
__global__ void __launch_bounds__(256)
cvt_bf16_k(const float* __restrict__ src, bf16_t* __restrict__ dst, long n)
{
  long idx = (long)blockIdx.x * 256 + threadIdx.x;
  if (idx < n) dst[idx] = (bf16_t)src[idx];
}

// src f32 [K,N]  -> dst bf16 transposed [NPad,KPad], zero padded
__global__ void __launch_bounds__(256)
convT_k(const float* __restrict__ src, bf16_t* __restrict__ dst,
        int K, int Nc, int KPad, long total)
{
  long idx = (long)blockIdx.x * 256 + threadIdx.x;
  if (idx >= total) return;
  int n = (int)(idx / KPad);
  int k = (int)(idx % KPad);
  dst[idx] = (k < K && n < Nc) ? (bf16_t)src[(size_t)k * Nc + n] : (bf16_t)0.f;
}

__global__ void __launch_bounds__(256)
addemb_k(float* __restrict__ h, const float* __restrict__ temb,
         const float* __restrict__ pos)
{
  long idx = (long)blockIdx.x * 256 + threadIdx.x;   // MTOK*DIMM
  int c = (int)(idx & 511);
  long m = idx >> 9;
  int b = (int)(m >> 11);
  int n = (int)(m & 2047);
  h[idx] += temb[b * DIMM + c] + pos[(size_t)n * DIMM + c];
}

__global__ void __launch_bounds__(256)
gelugate_k(const float* __restrict__ ff, bf16_t* __restrict__ ag)
{
  long idx = (long)blockIdx.x * 256 + threadIdx.x;   // MTOK * FFIPAD
  long m = idx / FFIPAD;
  int  c = (int)(idx % FFIPAD);
  float v = 0.f;
  if (c < FFI) {
    float u  = ff[m * FFI2PAD + c];
    float gt = ff[m * FFI2PAD + FFI + c];
    v = u * (0.5f * gt * (1.f + erff(gt * 0.70710678118654752f)));
  }
  ag[idx] = (bf16_t)v;
}

// ---------------------------------------------------------------------------
// host orchestration
// ---------------------------------------------------------------------------
extern "C" void kernel_launch(void* const* d_in, const int* in_sizes, int n_in,
                              void* d_out, int out_size, void* d_ws, size_t ws_size,
                              hipStream_t stream)
{
  (void)in_sizes; (void)n_in; (void)out_size; (void)ws_size;

  const float* X       = (const float*)d_in[0];
  const float* TIME    = (const float*)d_in[1];
  const float* POSE_W  = (const float*)d_in[2];
  const float* POSE_B  = (const float*)d_in[3];
  const float* POS_EMB = (const float*)d_in[4];
  const float* TW1 = (const float*)d_in[5];
  const float* TB1 = (const float*)d_in[6];
  const float* TW2 = (const float*)d_in[7];
  const float* TB2 = (const float*)d_in[8];
  const float* DW1 = (const float*)d_in[9];
  const float* DB1 = (const float*)d_in[10];
  const float* DW2 = (const float*)d_in[11];
  const float* DB2 = (const float*)d_in[12];
  const float* DW3 = (const float*)d_in[13];
  const float* DB3 = (const float*)d_in[14];
  const float* FIN_G    = (const float*)d_in[15];
  const float* FIN_B    = (const float*)d_in[16];
  const float* FIN_W    = (const float*)d_in[17];
  const float* FIN_BIAS = (const float*)d_in[18];
  auto LW = [&](int l, int j) { return (const float*)d_in[19 + l * 8 + j]; };
  // per layer: 0 ln1_g, 1 ln1_b, 2 wqkv, 3 wout, 4 ln2_g, 5 ln2_b, 6 ffw1, 7 ffw2

  char* base = (char*)d_ws;
  size_t off = 0;
  auto take = [&](size_t bytes) -> void* {
    void* p = (void*)(base + off);
    off += (bytes + 255) & ~(size_t)255;
    return p;
  };

  float*  H     = (float*) take((size_t)MTOK * DIMM * 4);
  bf16_t* XB    = (bf16_t*)take((size_t)MTOK * 64 * 2);
  float*  TEMB  = (float*) take((size_t)BATCH * DIMM * 4);
  float*  BALL  = (float*) take((size_t)512 * HEADS * 4);
  float*  BIAS  = (float*) take((size_t)HEADS * WWIN * 512 * 4);
  bf16_t* AIN   = (bf16_t*)take((size_t)MTOK * DIMM * 2);
  bf16_t* QKV   = (bf16_t*)take((size_t)MTOK * 3 * DIMM * 2);
  bf16_t* VT    = (bf16_t*)take((size_t)BATCH * HEADS * DH * SEQ * 2);
  bf16_t* OBUF  = (bf16_t*)take((size_t)MTOK * DIMM * 2);
  float*  FF    = (float*) take((size_t)MTOK * FFI2PAD * 4);
  bf16_t* AG    = (bf16_t*)take((size_t)MTOK * FFIPAD * 2);
  bf16_t* POSET = (bf16_t*)take((size_t)DIMM * 64 * 2);
  bf16_t* FINT  = (bf16_t*)take((size_t)64 * DIMM * 2);
  bf16_t *WQKVT[4], *WOUTT[4], *FFW1T[4], *FFW2T[4];
  for (int l = 0; l < 4; ++l) {
    WQKVT[l] = (bf16_t*)take((size_t)1536 * 512 * 2);
    WOUTT[l] = (bf16_t*)take((size_t)512 * 512 * 2);
    FFW1T[l] = (bf16_t*)take((size_t)FFI2PAD * 512 * 2);
    FFW2T[l] = (bf16_t*)take((size_t)512 * FFIPAD * 2);
  }

  // ---- one-time conversions (deterministic, every launch) ----------------
  cvt_bf16_k<<<2048, 256, 0, stream>>>(X, XB, (long)MTOK * 64);
  convT_k<<<128, 256, 0, stream>>>(POSE_W, POSET, 64, 512, 64, 32768L);
  convT_k<<<128, 256, 0, stream>>>(FIN_W, FINT, 512, 64, 512, 32768L);
  for (int l = 0; l < 4; ++l) {
    convT_k<<<3072, 256, 0, stream>>>(LW(l, 2), WQKVT[l], 512, 1536, 512, 786432L);
    convT_k<<<1024, 256, 0, stream>>>(LW(l, 3), WOUTT[l], 512, 512, 512, 262144L);
    convT_k<<<5504, 256, 0, stream>>>(LW(l, 6), FFW1T[l], 512, FFI2, 512,
                                      (long)FFI2PAD * 512);
    convT_k<<<2752, 256, 0, stream>>>(LW(l, 7), FFW2T[l], FFI, 512, FFIPAD,
                                      (long)512 * FFIPAD);
  }

  time_k<<<BATCH, 256, 0, stream>>>(TIME, TW1, TB1, TW2, TB2, TEMB);
  dpb_k<<<512, 256, 0, stream>>>(DW1, DB1, DW2, DB2, DW3, DB3, BALL);
  dpbbias_k<<<4096, 256, 0, stream>>>(BALL, BIAS);

  // ---- embed: h = x @ pose_w + pose_b, then + temb + pos_emb -------------
  gemm_bf16_k<true, false, true, false, false><<<dim3(64, 8), 256, 0, stream>>>(
      XB, 64, POSET, 64, POSE_B, nullptr, 0, H, DIMM, nullptr, 0, nullptr,
      MTOK, 64, DIMM);
  addemb_k<<<16384, 256, 0, stream>>>(H, TEMB, POS_EMB);

  // ---- transformer layers ------------------------------------------------
  for (int l = 0; l < 4; ++l) {
    ln_k<<<2048, 128, 0, stream>>>(H, LW(l, 0), LW(l, 1), AIN, MTOK);
    gemm_bf16_k<false, false, false, true, true><<<dim3(64, 24), 256, 0, stream>>>(
        AIN, DIMM, WQKVT[l], 512, nullptr, nullptr, 0, nullptr, 0, QKV, 1536, VT,
        MTOK, 512, 1536);
    attn_k<<<1024, 128, 131072, stream>>>(QKV, VT, BIAS, OBUF);
    gemm_bf16_k<false, true, true, false, false><<<dim3(64, 8), 256, 0, stream>>>(
        OBUF, DIMM, WOUTT[l], 512, nullptr, H, DIMM, H, DIMM, nullptr, 0, nullptr,
        MTOK, 512, DIMM);
    ln_k<<<2048, 128, 0, stream>>>(H, LW(l, 4), LW(l, 5), AIN, MTOK);
    gemm_bf16_k<false, false, true, false, false><<<dim3(64, 43), 256, 0, stream>>>(
        AIN, DIMM, FFW1T[l], 512, nullptr, nullptr, 0, FF, FFI2PAD, nullptr, 0,
        nullptr, MTOK, 512, FFI2);
    gelugate_k<<<44032, 256, 0, stream>>>(FF, AG);
    gemm_bf16_k<false, true, true, false, false><<<dim3(64, 8), 256, 0, stream>>>(
        AG, FFIPAD, FFW2T[l], FFIPAD, nullptr, H, DIMM, H, DIMM, nullptr, 0,
        nullptr, MTOK, FFIPAD, DIMM);
  }

  // ---- final LN + projection --------------------------------------------
  ln_k<<<2048, 128, 0, stream>>>(H, FIN_G, FIN_B, AIN, MTOK);
  gemm_bf16_k<true, false, true, false, false><<<dim3(64, 1), 256, 0, stream>>>(
      AIN, DIMM, FINT, 512, FIN_BIAS, nullptr, 0, (float*)d_out, 64, nullptr, 0,
      nullptr, MTOK, 512, 64);
}
